// EOS_33036888440878
// MI455X (gfx1250) — compile-verified, matrix operand order checked by reading, not run on hardware
//
#include <hip/hip_runtime.h>
#include <hip/hip_bf16.h>

// ---------------------------------------------------------------------------
// Problem constants (match reference)
// ---------------------------------------------------------------------------
#define BB   4
#define NN   2048
#define DD   1024
#define KK   16
#define MM   (BB * NN)        // 8192 rows
#define TAU_INV 0.0625f
#define LN_EPS  1e-5f

typedef __attribute__((ext_vector_type(16))) __bf16 v16bf;
typedef __attribute__((ext_vector_type(8)))  __bf16 v8bf;
typedef __attribute__((ext_vector_type(8)))  float  v8f;
typedef __attribute__((ext_vector_type(4)))  unsigned int v4u;
typedef __attribute__((ext_vector_type(8)))  int v8i;
typedef __attribute__((ext_vector_type(4)))  int v4i;

#if defined(__has_builtin)
#  if __has_builtin(__builtin_amdgcn_tensor_load_to_lds) && \
      __has_builtin(__builtin_amdgcn_s_wait_tensorcnt)
#    define EOS_HAS_TDM 1
#  else
#    define EOS_HAS_TDM 0
#  endif
#else
#  define EOS_HAS_TDM 0
#endif

// GEMM tiling
#define TILE_K   128               // K chunk staged per TDM transfer
#define LDS_STR  136               // bf16 elems per LDS row (128 + 8 pad = 272B)

// ---------------------------------------------------------------------------
// f32 -> bf16 conversion (grid-stride)
// ---------------------------------------------------------------------------
__global__ void EOS_cvt_bf16(const float* __restrict__ src,
                             __bf16* __restrict__ dst, long n) {
    long i = (long)blockIdx.x * blockDim.x + threadIdx.x;
    long stride = (long)gridDim.x * blockDim.x;
    for (; i < n; i += stride) dst[i] = (__bf16)src[i];
}

// Pack [We(16,1024); Ws(16,1024); zeros(32,1024)] -> 64x1024 bf16
__global__ void EOS_pack_es(const float* __restrict__ We,
                            const float* __restrict__ Ws,
                            __bf16* __restrict__ dst) {
    long i = (long)blockIdx.x * blockDim.x + threadIdx.x;
    if (i >= 64L * DD) return;
    int row = (int)(i >> 10);
    int col = (int)(i & (DD - 1));
    float v = 0.f;
    if (row < 16)      v = We[row * DD + col];
    else if (row < 32) v = Ws[(row - 16) * DD + col];
    dst[i] = (__bf16)v;
}

// ---------------------------------------------------------------------------
// Gate activation: exp(logsigmoid(z)/16) computed stably in f32
// ---------------------------------------------------------------------------
__device__ __forceinline__ float EOS_fgate(float z) {
    float ls = (z >= 0.f) ? -log1pf(__expf(-z)) : (z - log1pf(__expf(z)));
    return __expf(ls * TAU_INV);
}

// Low 32 bits of a flat LDS address are the LDS byte offset (ISA 10.2).
__device__ __forceinline__ unsigned EOS_lds_off(const void* p) {
    return (unsigned)(unsigned long long)(uintptr_t)p;
}

// ---------------------------------------------------------------------------
// TDM: load a 64-row x 128-col bf16 tile (row-major, row stride = strd elems)
// from global into LDS at byte offset ldsb, padding each 256B LDS row by 16B
// (pad_interval code 5 = 64 DWORDs, pad_amount code 3 = 4 DWORDs) so the
// effective LDS row stride is 272B = LDS_STR bf16 elements.
// D# packing per CDNA5 ISA.
// ---------------------------------------------------------------------------
__device__ __forceinline__ void EOS_tdm_load_tile(unsigned ldsb,
                                                  unsigned long long ga,
                                                  unsigned td0, unsigned td1,
                                                  unsigned strd) {
#if EOS_HAS_TDM
    v4u g0;
    g0[0] = 1u;                                            // count=1, user desc
    g0[1] = ldsb;                                          // lds_addr
    g0[2] = (unsigned)ga;                                  // global_addr[31:0]
    g0[3] = (unsigned)((ga >> 32) & 0x01FFFFFFull)         // global_addr[56:32]
          | 0x80000000u;                                   // type=2 (bits 127:126)
    v8i g1;
    g1[0] = (int)((1u << 16)      // data_size = 2B
                | (1u << 20)      // pad_enable
                | (5u << 22)      // pad_interval: 64 DWORDs (256B)
                | (3u << 25));    // pad_amount:   4 DWORDs (16B)
    g1[1] = (int)((td0 & 0xFFFFu) << 16);                  // tensor_dim0[15:0]
    g1[2] = (int)(((td0 >> 16) & 0xFFFFu)                  // tensor_dim0[31:16]
                | ((td1 & 0xFFFFu) << 16));                // tensor_dim1[15:0]
    g1[3] = (int)(((td1 >> 16) & 0xFFFFu)                  // tensor_dim1[31:16]
                | ((unsigned)TILE_K << 16));               // tile_dim0 = 128
    g1[4] = 64;                                            // tile_dim1 = 64 rows
    g1[5] = (int)strd;                                     // tensor_dim0_stride
    g1[6] = 0;
    g1[7] = 0;
    v4i g2 = {0, 0, 0, 0};
    v4i g3 = {0, 0, 0, 0};
    v8i g4 = {0, 0, 0, 0, 0, 0, 0, 0};   // unused trailing group (6-arg form)
    __builtin_amdgcn_tensor_load_to_lds(g0, g1, g2, g3, g4, 0);
#else
    (void)ldsb; (void)ga; (void)td0; (void)td1; (void)strd;
#endif
}

// ---------------------------------------------------------------------------
// WMMA GEMM:  C[M x Nout] = A[M x Kd] * W[Nout x Kd]^T   (bf16 in, f32 out)
// Block: 256 threads = 8 waves stacked along M (tile 128M x 64N).
// B panel (64 x TILE_K) staged in LDS by the Tensor Data Mover, double
// buffered; waves read fragments with ds_load_b128.
// 16-bit WMMA operand layout (CDNA5 ISA 7.12.2):
//   lane l<16  : K elements {kb+0..7, kb+16..23}  of row/col l
//   lane l>=16 : K elements {kb+8..15, kb+24..31} of row/col l-16
// epi: 0 = plain store, 1 = gate activation epilogue.
// ---------------------------------------------------------------------------
__global__ __launch_bounds__(256)
void EOS_gemm_bf16(const __bf16* __restrict__ A,
                   const __bf16* __restrict__ W,
                   float* __restrict__ C,
                   int Kd, int Nout, int ldc, int epi) {
    const int lane = threadIdx.x & 31;
    const int wave = threadIdx.x >> 5;
    const int half = lane >> 4;       // 0 or 1
    const int l    = lane & 15;

    const int mrow = blockIdx.y * 128 + wave * 16;
    const int ncol = blockIdx.x * 64;

    __shared__ __bf16 ldsB[2][64 * LDS_STR];   // 2 x 17408 B

    const long arow = (long)(mrow + l) * Kd;
    const int  nchunks = Kd / TILE_K;

    v8f c0 = {}, c1 = {}, c2 = {}, c3 = {};

#if EOS_HAS_TDM
    // Prologue: stage chunk 0 into buffer 0.
    if (wave == 0) {
        unsigned long long ga =
            (unsigned long long)(uintptr_t)(W + (size_t)ncol * Kd);
        EOS_tdm_load_tile(EOS_lds_off(&ldsB[0][0]), ga,
                          (unsigned)Kd, (unsigned)(Nout - ncol), (unsigned)Kd);
        __builtin_amdgcn_s_wait_tensorcnt(0);
    }
    __syncthreads();
#endif

    for (int c = 0; c < nchunks; ++c) {
        const int bufc = c & 1;

#if EOS_HAS_TDM
        // Kick off the next chunk into the other buffer (its previous
        // readers finished at the end of iteration c-1).
        if (wave == 0 && c + 1 < nchunks) {
            unsigned long long ga = (unsigned long long)(uintptr_t)
                (W + (size_t)ncol * Kd + (size_t)(c + 1) * TILE_K);
            EOS_tdm_load_tile(EOS_lds_off(&ldsB[bufc ^ 1][0]), ga,
                              (unsigned)(Kd - (c + 1) * TILE_K),
                              (unsigned)(Nout - ncol), (unsigned)Kd);
        }
#else
        // Fallback: cooperative synchronous staging.
        __syncthreads();
        {
            const int t   = threadIdx.x;
            const int row = t >> 2;
            const int seg = (t & 3) * 32;
            const __bf16* src = W + (size_t)(ncol + row) * Kd
                                  + (size_t)c * TILE_K + seg;
            __bf16* dst = &ldsB[bufc][row * LDS_STR + seg];
            #pragma unroll
            for (int q = 0; q < 4; ++q)
                *(v8bf*)(dst + q * 8) = *(const v8bf*)(src + q * 8);
        }
        __syncthreads();
#endif

        if (c + 1 < nchunks)   // stream A ahead (-> global_prefetch_b8)
            __builtin_prefetch(A + arow + (c + 1) * TILE_K, 0, 3);

        const __bf16* lb = &ldsB[bufc][0];
        #pragma unroll
        for (int ks = 0; ks < 4; ++ks) {
            const int kg = c * TILE_K + ks * 32;   // global K base
            const int kl = ks * 32 + half * 8;     // local K offset in chunk

            union { v16bf v; v8bf h[2]; } af, b0, b1, b2, b3;
            af.h[0] = *(const v8bf*)(A + arow + kg + half * 8);
            af.h[1] = *(const v8bf*)(A + arow + kg + 16 + half * 8);

            b0.h[0] = *(const v8bf*)(lb + (l +  0) * LDS_STR + kl);
            b0.h[1] = *(const v8bf*)(lb + (l +  0) * LDS_STR + kl + 16);
            b1.h[0] = *(const v8bf*)(lb + (l + 16) * LDS_STR + kl);
            b1.h[1] = *(const v8bf*)(lb + (l + 16) * LDS_STR + kl + 16);
            b2.h[0] = *(const v8bf*)(lb + (l + 32) * LDS_STR + kl);
            b2.h[1] = *(const v8bf*)(lb + (l + 32) * LDS_STR + kl + 16);
            b3.h[0] = *(const v8bf*)(lb + (l + 48) * LDS_STR + kl);
            b3.h[1] = *(const v8bf*)(lb + (l + 48) * LDS_STR + kl + 16);

            c0 = __builtin_amdgcn_wmma_f32_16x16x32_bf16(false, af.v, false, b0.v,
                                                         (short)0, c0, false, false);
            c1 = __builtin_amdgcn_wmma_f32_16x16x32_bf16(false, af.v, false, b1.v,
                                                         (short)0, c1, false, false);
            c2 = __builtin_amdgcn_wmma_f32_16x16x32_bf16(false, af.v, false, b2.v,
                                                         (short)0, c2, false, false);
            c3 = __builtin_amdgcn_wmma_f32_16x16x32_bf16(false, af.v, false, b3.v,
                                                         (short)0, c3, false, false);
        }

#if EOS_HAS_TDM
        __syncthreads();                       // everyone done reading bufc
        if (c + 1 < nchunks) {
            if (wave == 0) __builtin_amdgcn_s_wait_tensorcnt(0);
            __syncthreads();                   // next buffer visible to all
        }
#endif
    }

    // C layout: VGPR v, lane l<16 -> (M = v,   N = l)
    //                    lane>=16 -> (M = v+8, N = l-16)
    #pragma unroll
    for (int v = 0; v < 8; ++v) {
        long roff = (long)(mrow + half * 8 + v) * ldc + ncol + l;
        float x0 = c0[v], x1 = c1[v], x2 = c2[v], x3 = c3[v];
        if (epi == 1) {
            x0 = EOS_fgate(x0); x1 = EOS_fgate(x1);
            x2 = EOS_fgate(x2); x3 = EOS_fgate(x3);
        }
        C[roff +  0] = x0;
        C[roff + 16] = x1;
        C[roff + 32] = x2;
        C[roff + 48] = x3;
    }
}

// ---------------------------------------------------------------------------
// Sequential gated scan over N.
// grid = 64 blocks (b in [0,4) x d-chunk in [0,16)), block = 64 threads.
// Thread owns one (b, d): 16-element state vector m[k] in registers.
//   m[k] = f[t,d]*m[k] + e[t,k]*i[t,d];   y[t,d] = sum_k s[t,k]*m[k]
// e/s staged through LDS in chunks of 64 timesteps; per-step f/i global
// loads software-pipelined (t+1 issued before the FMA chain of t).
// ---------------------------------------------------------------------------
__global__ __launch_bounds__(64)
void EOS_scan(const float* __restrict__ ibuf,   // [MM, DD]
              const float* __restrict__ fbuf,   // [MM, DD]
              const float* __restrict__ esf,    // [MM, 64] (e:0..15, s:16..31)
              float* __restrict__ ybuf) {       // [MM, DD]
    const int b     = blockIdx.x >> 4;
    const int dbase = (blockIdx.x & 15) * 64;
    const int tid   = threadIdx.x;
    const int d     = dbase + tid;
    const long rowB = (long)b * NN;

    __shared__ float es[64 * 32];   // 8 KB

    float m[KK];
    #pragma unroll
    for (int k = 0; k < KK; ++k) m[k] = 0.f;

    float fv = fbuf[rowB * DD + d];
    float iv = ibuf[rowB * DD + d];

    for (int t0 = 0; t0 < NN; t0 += 64) {
        __syncthreads();
        {   // thread tid stages the 32 e/s floats of timestep t0+tid
            const float4* src = (const float4*)&esf[(rowB + t0 + tid) * 64];
            float4* dst = (float4*)&es[tid * 32];
            #pragma unroll
            for (int q = 0; q < 8; ++q) dst[q] = src[q];
        }
        __syncthreads();

        for (int tt = 0; tt < 64; ++tt) {
            const int  t   = t0 + tt;
            const long row = rowB + t;
            float fn = 0.f, in2 = 0.f;
            if (t + 1 < NN) {                  // pipeline next step's loads
                fn  = fbuf[(row + 1) * DD + d];
                in2 = ibuf[(row + 1) * DD + d];
            }
            float y = 0.f;
            #pragma unroll
            for (int k = 0; k < KK; ++k) {
                m[k] = fmaf(fv, m[k], es[tt * 32 + k] * iv);
                y    = fmaf(es[tt * 32 + 16 + k], m[k], y);
            }
            ybuf[row * DD + d] = y;
            fv = fn; iv = in2;
        }
    }
}

// ---------------------------------------------------------------------------
// LayerNorm over D=1024, output bf16. One block (256 thr) per row.
// ---------------------------------------------------------------------------
__global__ __launch_bounds__(256)
void EOS_layernorm(const float* __restrict__ Y,
                   const float* __restrict__ gamma,
                   const float* __restrict__ beta,
                   __bf16* __restrict__ out) {
    const int r   = blockIdx.x;
    const int tid = threadIdx.x;
    const float* y = Y + (long)r * DD;

    float4 v = ((const float4*)y)[tid];
    float s  = v.x + v.y + v.z + v.w;
    float s2 = v.x * v.x + v.y * v.y + v.z * v.z + v.w * v.w;

    __shared__ float r1[256];
    __shared__ float r2[256];
    r1[tid] = s; r2[tid] = s2;
    __syncthreads();
    for (int off = 128; off > 0; off >>= 1) {
        if (tid < off) { r1[tid] += r1[tid + off]; r2[tid] += r2[tid + off]; }
        __syncthreads();
    }
    const float mu  = r1[0] * (1.f / DD);
    const float var = r2[0] * (1.f / DD) - mu * mu;
    const float rs  = rsqrtf(var + LN_EPS);

    const int d0 = tid * 4;
    float4 g  = ((const float4*)gamma)[tid];
    float4 bt = ((const float4*)beta)[tid];
    __bf16* o = out + (long)r * DD + d0;
    o[0] = (__bf16)((v.x - mu) * rs * g.x + bt.x);
    o[1] = (__bf16)((v.y - mu) * rs * g.y + bt.y);
    o[2] = (__bf16)((v.z - mu) * rs * g.z + bt.z);
    o[3] = (__bf16)((v.w - mu) * rs * g.w + bt.w);
}

// ---------------------------------------------------------------------------
// Host-side launcher
// ---------------------------------------------------------------------------
extern "C" void kernel_launch(void* const* d_in, const int* in_sizes, int n_in,
                              void* d_out, int out_size, void* d_ws, size_t ws_size,
                              hipStream_t stream) {
    (void)in_sizes; (void)n_in; (void)out_size; (void)ws_size;

    const float* x     = (const float*)d_in[0];   // [B,N,D]
    const float* Wi    = (const float*)d_in[1];   // [D,D]
    const float* We    = (const float*)d_in[2];   // [K,D]
    const float* Wf    = (const float*)d_in[3];   // [D,D]
    const float* Ws    = (const float*)d_in[4];   // [K,D]
    const float* gamma = (const float*)d_in[5];   // [D]
    const float* beta  = (const float*)d_in[6];   // [D]
    const float* Wo    = (const float*)d_in[7];   // [D,D]

    // ---- carve workspace -------------------------------------------------
    char* p = (char*)d_ws;
    auto carve = [&](size_t bytes) {
        char* q = p;
        p += (bytes + 255) & ~(size_t)255;
        return (void*)q;
    };
    __bf16* xb   = (__bf16*)carve((size_t)MM * DD * 2);   // x in bf16
    __bf16* Wib  = (__bf16*)carve((size_t)DD * DD * 2);
    __bf16* Wfb  = (__bf16*)carve((size_t)DD * DD * 2);
    __bf16* Wob  = (__bf16*)carve((size_t)DD * DD * 2);
    __bf16* Wesb = (__bf16*)carve((size_t)64 * DD * 2);   // [We;Ws;pad]
    float*  ibuf = (float*)carve((size_t)MM * DD * 4);
    float*  fbuf = (float*)carve((size_t)MM * DD * 4);
    float*  esf  = (float*)carve((size_t)MM * 64 * 4);
    float*  ybuf = (float*)carve((size_t)MM * DD * 4);
    __bf16* ynb  = (__bf16*)carve((size_t)MM * DD * 2);

    // ---- 1. precision conversion ----------------------------------------
    EOS_cvt_bf16<<<4096, 256, 0, stream>>>(x,  xb,  (long)MM * DD);
    EOS_cvt_bf16<<<1024, 256, 0, stream>>>(Wi, Wib, (long)DD * DD);
    EOS_cvt_bf16<<<1024, 256, 0, stream>>>(Wf, Wfb, (long)DD * DD);
    EOS_cvt_bf16<<<1024, 256, 0, stream>>>(Wo, Wob, (long)DD * DD);
    EOS_pack_es<<<(64 * DD + 255) / 256, 256, 0, stream>>>(We, Ws, Wesb);

    // ---- 2. projections (WMMA bf16, TDM-staged B panels) -----------------
    dim3 blk(256);
    dim3 grdBig(DD / 64, MM / 128);     // (16, 64)
    dim3 grdES(1, MM / 128);            // (1, 64)
    EOS_gemm_bf16<<<grdBig, blk, 0, stream>>>(xb, Wib,  ibuf, DD, DD, DD, 0);
    EOS_gemm_bf16<<<grdBig, blk, 0, stream>>>(xb, Wfb,  fbuf, DD, DD, DD, 1);
    EOS_gemm_bf16<<<grdES,  blk, 0, stream>>>(xb, Wesb, esf,  DD, 64, 64, 0);

    // ---- 3. sequential gated scan ---------------------------------------
    EOS_scan<<<64, 64, 0, stream>>>(ibuf, fbuf, esf, ybuf);

    // ---- 4. LayerNorm -> bf16 -------------------------------------------
    EOS_layernorm<<<MM, 256, 0, stream>>>(ybuf, gamma, beta, ynb);

    // ---- 5. output projection (WMMA bf16) -> d_out (f32) ----------------
    EOS_gemm_bf16<<<grdBig, blk, 0, stream>>>(ynb, Wob, (float*)d_out,
                                              DD, DD, DD, 0);
}
